// NACVQVAE_79156247265421
// MI455X (gfx1250) — matrix operand order repaired and checked by reading, not compiled, not run on hardware
//
#include <hip/hip_runtime.h>

typedef __attribute__((ext_vector_type(16))) __bf16 v16bf;
typedef __attribute__((ext_vector_type(8)))  float  v8f;

#define B_SZ 16
#define CE   1024
#define T_SZ 2048
#define KCB  4096
#define CC   64

// workspace byte offsets
#define WINA_OFF   0u         // 128KB  W_in A-fragments  [32 kstep][4 mt][32 lane][16 bf16]
#define WOUTA_OFF  131072u    // 128KB  W_out A-fragments [64 mt][32 lane][32 bf16] (k2*16+e)
#define CBA_OFF    262144u    // 512KB  cb_n A-fragments  [256 mt][32 lane][32 bf16]
#define CBN2_OFF   786432u    // 16KB   ||cb_n||^2
#define LOSS_OFF   802816u    // 4B     loss accumulator
#define EN_OFF     1048576u   // 4MB    e normalized bf16 [b][t][64]
#define QBF_OFF    5242880u   // 4MB    q bf16            [b][t][64]

// d_out float offsets (codes, quantized, cb_loss, commit_loss, x_proj, quantized_proj)
#define O_CODES 0
#define O_QUANT 32768
#define O_CBL   33587200
#define O_CML   33587201
#define O_XPROJ 33587202
#define O_QPROJ 35684354

__device__ __forceinline__ v8f wmma_bf16(v16bf a, v16bf b, v8f c) {
  return __builtin_amdgcn_wmma_f32_16x16x32_bf16(false, a, false, b, (short)0, c, false, false);
}

// cooperative 32KB global->LDS chunk copy (256 threads, 128B each)
__device__ __forceinline__ void block_copy_32k(uint4* l, const uint4* g) {
#pragma unroll
  for (int i = 0; i < 8; ++i) l[threadIdx.x + 256 * i] = g[threadIdx.x + 256 * i];
}

// ---- A-fragment slot math (16-bit A 16x32, contiguous k order chosen by this file; B uses the same) ----
// lane = (M%16) + 16*h ; element e: K%32 = 16h + e

// Prep: W_in = g*v/||v|| into fragment layout; also zero loss accumulator.
__global__ void prep_win(const float* __restrict__ v_in, const float* __restrict__ g_in,
                         __bf16* __restrict__ winA, float* __restrict__ loss) {
  int o = blockIdx.x, j = threadIdx.x;
  __shared__ float red[256];
  float4 v = reinterpret_cast<const float4*>(v_in + o * CE)[j];
  red[j] = v.x * v.x + v.y * v.y + v.z * v.z + v.w * v.w;
  __syncthreads();
  for (int st = 128; st > 0; st >>= 1) { if (j < st) red[j] += red[j + st]; __syncthreads(); }
  float scale = g_in[o] * rsqrtf(red[0]);
  if (o == 0 && j == 0) *loss = 0.0f;
  int mt = o >> 4;
  float vv[4] = {v.x, v.y, v.z, v.w};
#pragma unroll
  for (int kk = 0; kk < 4; ++kk) {
    int Kv = j * 4 + kk;
    int kstep = Kv >> 5, w32 = Kv & 31;
    int h = (w32 >> 4) & 1, base = w32 - 16 * h;
    int e = base;
    int lane = (o & 15) + 16 * h;
    winA[((kstep * 4 + mt) * 32 + lane) * 16 + e] = (__bf16)(vv[kk] * scale);
  }
}

__global__ void prep_wout(const float* __restrict__ v_out, const float* __restrict__ g_out,
                          __bf16* __restrict__ woutA) {
  int o = blockIdx.x, c = threadIdx.x;
  __shared__ float red[64];
  float v = v_out[o * CC + c];
  red[c] = v * v; __syncthreads();
  for (int st = 32; st > 0; st >>= 1) { if (c < st) red[c] += red[c + st]; __syncthreads(); }
  float scale = g_out[o] * rsqrtf(red[0]);
  int mt = o >> 4;
  int k2 = c >> 5, w32 = c & 31;
  int h = (w32 >> 4) & 1, base = w32 - 16 * h;
  int e = base;
  int lane = (o & 15) + 16 * h;
  woutA[(mt * 32 + lane) * 32 + k2 * 16 + e] = (__bf16)(v * scale);
}

__global__ void prep_cb(const float* __restrict__ codebook, __bf16* __restrict__ cbA,
                        float* __restrict__ cbn2) {
  int k = blockIdx.x, c = threadIdx.x;
  __shared__ float red[64];
  float v = codebook[k * CC + c];
  red[c] = v * v; __syncthreads();
  for (int st = 32; st > 0; st >>= 1) { if (c < st) red[c] += red[c + st]; __syncthreads(); }
  float sumsq = red[0];
  float inv = 1.0f / fmaxf(sqrtf(sumsq), 1e-12f);
  if (c == 0) cbn2[k] = sumsq * inv * inv;
  int mt = k >> 4;
  int k2 = c >> 5, w32 = c & 31;
  int h = (w32 >> 4) & 1, base = w32 - 16 * h;
  int e = base;
  int lane = (k & 15) + 16 * h;
  cbA[(mt * 32 + lane) * 32 + k2 * 16 + e] = (__bf16)(v * inv);
}

// k1: x_proj[b] (64 x T) = W_in @ x[b]; per-column L2 norm; write fp32 x_proj + bf16 e[t][c].
// One wave: 32 t (2 N-tiles), M=64 (4 tiles), K=1024 (32 steps), W_in staged via LDS chunks.
__global__ void __launch_bounds__(256) k1_proj_norm(const float* __restrict__ x,
                                                    const __bf16* __restrict__ winA,
                                                    float* __restrict__ xproj_out,
                                                    __bf16* __restrict__ en_out) {
  __shared__ __align__(32) __bf16 lds_a[16384];          // 8 ksteps * 4 mt * 32 lane * 16
  __shared__ __align__(16) __bf16 lds_en[8][16][64];
  int warp = threadIdx.x >> 5, lane = threadIdx.x & 31;
  int w = blockIdx.x * 8 + warp;
  int b = w >> 6;
  int t0 = (w & 63) * 32;
  int h = lane >> 4, n = lane & 15;

  v8f acc[4][2];
#pragma unroll
  for (int mt = 0; mt < 4; ++mt)
#pragma unroll
    for (int nt = 0; nt < 2; ++nt)
      acc[mt][nt] = v8f{0.f, 0.f, 0.f, 0.f, 0.f, 0.f, 0.f, 0.f};

  const float* xb = x + (size_t)b * CE * T_SZ;
  for (int ch = 0; ch < 4; ++ch) {
    __syncthreads();
    block_copy_32k(reinterpret_cast<uint4*>(lds_a),
                   reinterpret_cast<const uint4*>(winA + ch * 16384));
    __syncthreads();
    if (ch < 3) __builtin_prefetch(winA + (ch + 1) * 16384 + threadIdx.x * 64, 0, 3);
    for (int ks = 0; ks < 8; ++ks) {
      int kstep = ch * 8 + ks;
      union { v16bf v; __bf16 s[16]; } bu[2];
#pragma unroll
      for (int nt = 0; nt < 2; ++nt) {
        int t = t0 + nt * 16 + n;
#pragma unroll
        for (int e = 0; e < 16; ++e) {  // B 32x16 layout: element e -> K = 32*kstep + 16h + e
          int Kv = kstep * 32 + 16 * h + e;
          bu[nt].s[e] = (__bf16)xb[Kv * T_SZ + t];
        }
      }
#pragma unroll
      for (int mt = 0; mt < 4; ++mt) {
        v16bf a = *reinterpret_cast<const v16bf*>(lds_a + ((ks * 4 + mt) * 32 + lane) * 16);
        acc[mt][0] = wmma_bf16(a, bu[0].v, acc[mt][0]);
        acc[mt][1] = wmma_bf16(a, bu[1].v, acc[mt][1]);
      }
    }
  }

  for (int nt = 0; nt < 2; ++nt) {
    int t = t0 + nt * 16 + n;
    float ss = 0.f;
#pragma unroll
    for (int mt = 0; mt < 4; ++mt)
#pragma unroll
      for (int r = 0; r < 8; ++r) { float vv = acc[mt][nt][r]; ss += vv * vv; }
    ss += __shfl_xor(ss, 16, 32);                 // combine the two lane-halves of this column
    float inv = 1.0f / fmaxf(sqrtf(ss), 1e-12f);
#pragma unroll
    for (int mt = 0; mt < 4; ++mt)
#pragma unroll
      for (int r = 0; r < 8; ++r) {
        int m = mt * 16 + 8 * h + r;
        float vv = acc[mt][nt][r];
        xproj_out[(b * CC + m) * T_SZ + t] = vv;
        lds_en[warp][n][m] = (__bf16)(vv * inv);
      }
    asm volatile("s_wait_dscnt 0" ::: "memory");  // cross-lane LDS visibility within wave
    {
      int row = lane >> 1, hs = lane & 1;
      const uint4* s = reinterpret_cast<const uint4*>(&lds_en[warp][row][hs * 32]);
      uint4* d = reinterpret_cast<uint4*>(en_out + ((b * T_SZ + t0 + nt * 16 + row) * 64 + hs * 32));
#pragma unroll
      for (int q = 0; q < 4; ++q) d[q] = s[q];
    }
    asm volatile("s_wait_dscnt 0" ::: "memory");
  }
}

// k2: per 16-t tile: dot = cb_n (4096x64) @ e (64x16); argmin(cbn2 - 2*dot); gather; losses.
// cb_n fragments staged through LDS in 16-mt (32KB) chunks shared by all 8 waves.
__global__ void __launch_bounds__(256) k2_argmin(const __bf16* __restrict__ en,
                                                 const __bf16* __restrict__ cbA,
                                                 const float* __restrict__ cbn2,
                                                 const float* __restrict__ codebook,
                                                 const float* __restrict__ xproj,
                                                 float* __restrict__ codes_out,
                                                 float* __restrict__ qproj_out,
                                                 __bf16* __restrict__ qbf_out,
                                                 float* __restrict__ loss) {
  __shared__ __align__(32) __bf16 lds_a[16384];          // 16 mt * 32 lane * 32
  __shared__ __align__(16) __bf16 lds_q[8][16][64];
  int warp = threadIdx.x >> 5, lane = threadIdx.x & 31;
  int w = blockIdx.x * 8 + warp;
  int b = w >> 7;
  int t0 = (w & 127) * 16;
  int h = lane >> 4, n = lane & 15;
  int t = t0 + n;

  v16bf b0 = *reinterpret_cast<const v16bf*>(en + (b * T_SZ + t) * 64 + 16 * h);
  v16bf b1 = *reinterpret_cast<const v16bf*>(en + (b * T_SZ + t) * 64 + 32 + 16 * h);

  float best = 3.4e38f; int bestk = 0;
  for (int ch = 0; ch < 16; ++ch) {
    __syncthreads();
    block_copy_32k(reinterpret_cast<uint4*>(lds_a),
                   reinterpret_cast<const uint4*>(cbA + ch * 16384));
    __syncthreads();
    if (ch < 15) __builtin_prefetch(cbA + (ch + 1) * 16384 + threadIdx.x * 64, 0, 3);
    for (int mtl = 0; mtl < 16; ++mtl) {
      int mt = ch * 16 + mtl;
      const __bf16* ap = lds_a + (mtl * 32 + lane) * 32;
      v16bf a0 = *reinterpret_cast<const v16bf*>(ap);
      v16bf a1 = *reinterpret_cast<const v16bf*>(ap + 16);
      v8f acc = v8f{0.f, 0.f, 0.f, 0.f, 0.f, 0.f, 0.f, 0.f};
      acc = wmma_bf16(a0, b0, acc);
      acc = wmma_bf16(a1, b1, acc);
      const float4* c2p = reinterpret_cast<const float4*>(cbn2 + mt * 16 + 8 * h);
      float4 c2a = c2p[0], c2b = c2p[1];
      float c2v[8] = {c2a.x, c2a.y, c2a.z, c2a.w, c2b.x, c2b.y, c2b.z, c2b.w};
#pragma unroll
      for (int r = 0; r < 8; ++r) {
        float score = c2v[r] - 2.0f * acc[r];
        int kk = mt * 16 + 8 * h + r;
        if (score < best) { best = score; bestk = kk; }
      }
    }
  }
  float ob = __shfl_xor(best, 16, 32);
  int obk = __shfl_xor(bestk, 16, 32);
  if (ob < best || (ob == best && obk < bestk)) { best = ob; bestk = obk; }
  if (h == 0) codes_out[b * T_SZ + t] = (float)bestk;

  // gather un-normalized codebook row, loss partials, q outputs; lane covers c in [32h, 32h+32)
  float lsum = 0.f;
  const float4* cbrow = reinterpret_cast<const float4*>(codebook + bestk * CC + 32 * h);
#pragma unroll
  for (int j = 0; j < 8; ++j) {
    float4 q4 = cbrow[j];
    float qv[4] = {q4.x, q4.y, q4.z, q4.w};
#pragma unroll
    for (int i = 0; i < 4; ++i) {
      int c = 32 * h + j * 4 + i;
      float xp = xproj[(b * CC + c) * T_SZ + t];
      float d = xp - qv[i];
      lsum += d * d;
      qproj_out[(b * CC + c) * T_SZ + t] = qv[i];
      lds_q[warp][n][c] = (__bf16)qv[i];
    }
  }
#pragma unroll
  for (int m = 1; m < 32; m <<= 1) lsum += __shfl_xor(lsum, m, 32);
  if (lane == 0) atomicAdd(loss, lsum);

  asm volatile("s_wait_dscnt 0" ::: "memory");
  {
    int row = lane >> 1, hs = lane & 1;
    const uint4* s = reinterpret_cast<const uint4*>(&lds_q[warp][row][hs * 32]);
    uint4* d = reinterpret_cast<uint4*>(qbf_out + ((b * T_SZ + t0 + row) * 64 + hs * 32));
#pragma unroll
    for (int q = 0; q < 4; ++q) d[q] = s[q];
  }
}

// k3: quantized[b] (1024 x T) = W_out @ q[b]; M=1024 (64 tiles), K=64, 32 t per wave.
__global__ void __launch_bounds__(256) k3_out(const __bf16* __restrict__ qbf,
                                              const __bf16* __restrict__ woutA,
                                              float* __restrict__ quant_out) {
  __shared__ __align__(32) __bf16 lds_a[16384];          // 16 mt * 32 lane * 32
  int warp = threadIdx.x >> 5, lane = threadIdx.x & 31;
  int w = blockIdx.x * 8 + warp;
  int b = w >> 6;
  int t0 = (w & 63) * 32;
  int h = lane >> 4, n = lane & 15;

  v16bf bf[2][2];
#pragma unroll
  for (int nt = 0; nt < 2; ++nt) {
    int t = t0 + nt * 16 + n;
    bf[nt][0] = *reinterpret_cast<const v16bf*>(qbf + (b * T_SZ + t) * 64 + 16 * h);
    bf[nt][1] = *reinterpret_cast<const v16bf*>(qbf + (b * T_SZ + t) * 64 + 32 + 16 * h);
  }
  for (int ch = 0; ch < 4; ++ch) {
    __syncthreads();
    block_copy_32k(reinterpret_cast<uint4*>(lds_a),
                   reinterpret_cast<const uint4*>(woutA + ch * 16384));
    __syncthreads();
    if (ch < 3) __builtin_prefetch(woutA + (ch + 1) * 16384 + threadIdx.x * 64, 0, 3);
    for (int mtl = 0; mtl < 16; ++mtl) {
      int mt = ch * 16 + mtl;
      const __bf16* ap = lds_a + (mtl * 32 + lane) * 32;
      v16bf a0 = *reinterpret_cast<const v16bf*>(ap);
      v16bf a1 = *reinterpret_cast<const v16bf*>(ap + 16);
#pragma unroll
      for (int nt = 0; nt < 2; ++nt) {
        v8f acc = v8f{0.f, 0.f, 0.f, 0.f, 0.f, 0.f, 0.f, 0.f};
        acc = wmma_bf16(a0, bf[nt][0], acc);
        acc = wmma_bf16(a1, bf[nt][1], acc);
        int t = t0 + nt * 16 + n;
#pragma unroll
        for (int r = 0; r < 8; ++r) {
          int m = mt * 16 + 8 * h + r;
          quant_out[(b * CE + m) * T_SZ + t] = acc[r];
        }
      }
    }
  }
}

__global__ void k_final(const float* __restrict__ loss, float* __restrict__ out) {
  if (threadIdx.x == 0) {
    float m = loss[0] / (float)(B_SZ * CC * T_SZ);
    out[O_CBL] = m;   // codebook_loss == commit_loss numerically (stop_gradient is identity fwd)
    out[O_CML] = m;
  }
}

extern "C" void kernel_launch(void* const* d_in, const int* in_sizes, int n_in,
                              void* d_out, int out_size, void* d_ws, size_t ws_size,
                              hipStream_t stream) {
  const float* x        = (const float*)d_in[0];
  const float* v_in     = (const float*)d_in[1];
  const float* g_in     = (const float*)d_in[2];
  const float* v_out    = (const float*)d_in[3];
  const float* g_out    = (const float*)d_in[4];
  const float* codebook = (const float*)d_in[5];
  float* out = (float*)d_out;
  char*  ws  = (char*)d_ws;

  __bf16* winA  = (__bf16*)(ws + WINA_OFF);
  __bf16* woutA = (__bf16*)(ws + WOUTA_OFF);
  __bf16* cbA   = (__bf16*)(ws + CBA_OFF);
  float*  cbn2  = (float*)(ws + CBN2_OFF);
  float*  loss  = (float*)(ws + LOSS_OFF);
  __bf16* en    = (__bf16*)(ws + EN_OFF);
  __bf16* qbf   = (__bf16*)(ws + QBF_OFF);

  prep_win <<<64,   256, 0, stream>>>(v_in,  g_in,  winA, loss);
  prep_wout<<<1024, 64,  0, stream>>>(v_out, g_out, woutA);
  prep_cb  <<<4096, 64,  0, stream>>>(codebook, cbA, cbn2);

  k1_proj_norm<<<128, 256, 0, stream>>>(x, winA, out + O_XPROJ, en);
  k2_argmin   <<<256, 256, 0, stream>>>(en, cbA, cbn2, codebook, out + O_XPROJ,
                                        out + O_CODES, out + O_QPROJ, qbf, loss);
  k3_out      <<<128, 256, 0, stream>>>(qbf, woutA, out + O_QUANT);
  k_final     <<<1,   32,  0, stream>>>(loss, out);
}
